// ViT_66571993088878
// MI455X (gfx1250) — matrix operand order repaired and checked by reading, not compiled, not run on hardware
//
#include <hip/hip_runtime.h>
#include <hip/hip_bf16.h>

// ---------------------------------------------------------------------------
// MI455X (gfx1250) ViT pipeline, round 4.
//  - kg_gemm: LDS-free, branch-free WMMA GEMM; 16x32 tile per wave
//    (1 A fragment reused across 2 B fragments -> 2 wmma per K-step).
//  - conv2d: A direct from global (padded weights), 32-pixel B panel in LDS,
//    2 wmma per K-step; X1 halo-padded so staging has no masks.
//  - attention q/k/v tiles via global_load_async_to_lds_b32 + s_wait_asynccnt.
// ---------------------------------------------------------------------------

typedef __attribute__((ext_vector_type(16))) _Float16 v16h;
typedef __attribute__((ext_vector_type(8)))  _Float16 h8;
typedef __attribute__((ext_vector_type(8)))  float    v8f;

#define Bsz   128
#define IMGD  25
#define Nn    25
#define DIMd  128
#define HEADS 8
#define DHh   64
#define INNER 512
#define MLPd  256
#define NCc   16
#define EPSf  1e-5f
#define IC2   672          // 24 * 28
#define K2    6048         // 672 * 9
#define PATCH 750          // 5*5*30
#define PATCHP 768         // padded K for WMMA
#define TOK   3200         // B * N
#define XPD   27           // halo-padded conv3d output H/W
#define XPD2  (XPD * XPD)  // 729

static __device__ __forceinline__ float frelu(float v) { return v > 0.f ? v : 0.f; }

#define SHUF16(lo, hi) __builtin_shufflevector(lo, hi, 0,1,2,3,4,5,6,7,8,9,10,11,12,13,14,15)
#define WMMA16(af, bf, acc) __builtin_amdgcn_wmma_f32_16x16x32_f16(false, af, false, bf, (short)0, acc, false, false)

// async global -> LDS, 4 bytes per active lane (tracked by ASYNCcnt)
static __device__ __forceinline__ void async_ld32(void* lds, const void* gaddr) {
    uint32_t ldsOff = (uint32_t)(uintptr_t)lds;   // generic LDS addr: low 32 = offset
    asm volatile("global_load_async_to_lds_b32 %0, %1, off"
                 :: "v"(ldsOff), "v"(gaddr) : "memory");
}
static __device__ __forceinline__ void async_wait0() {
    asm volatile("s_wait_asynccnt 0x0" ::: "memory");
}

// ---------------------------------------------------------------------------
// zero fp16 buffer (vectorized, n % 8 == 0)
// ---------------------------------------------------------------------------
__global__ void k_zero_h(_Float16* __restrict__ p, int n8) {
    int i = blockIdx.x * 256 + threadIdx.x;
    if (i < n8) { h8 z = {}; *(h8*)(p + (size_t)i * 8) = z; }
}

// ---------------------------------------------------------------------------
// fp32 [K][N] -> fp16 transposed+padded [N][Kp]
// ---------------------------------------------------------------------------
__global__ void k_f2h_t(const float* __restrict__ W, _Float16* __restrict__ BT,
                        int K, int N, int Kp) {
    int i = blockIdx.x * 256 + threadIdx.x;
    if (i >= N * Kp) return;
    int n = i / Kp, k = i % Kp;
    BT[i] = (k < K) ? (_Float16)W[(size_t)k * N + n] : (_Float16)0.f;
}

// conv2d weights fp32 [30][6048] -> fp16 [32][6048], rows 30..31 zero
__global__ void k_prep_w2(const float* __restrict__ W, _Float16* __restrict__ Wh) {
    int i = blockIdx.x * 256 + threadIdx.x;
    if (i >= 32 * K2) return;
    int oc = i / K2;
    Wh[i] = (oc < 30) ? (_Float16)W[i] : (_Float16)0.f;
}

// ---------------------------------------------------------------------------
// conv3d (no D-pad, HW pad 1) + BN + ReLU -> fp16 halo-padded [B][672][27][27]
// (halo ring pre-zeroed by k_zero_h)
// ---------------------------------------------------------------------------
__global__ __launch_bounds__(256) void k_conv3d(
    const float* __restrict__ img, const float* __restrict__ w3,
    const float* __restrict__ b3, const float* __restrict__ g,
    const float* __restrict__ bb, const float* __restrict__ mu,
    const float* __restrict__ var, _Float16* __restrict__ X1)
{
    int idx = blockIdx.x * 256 + threadIdx.x;
    if (idx >= Bsz * IC2 * IMGD * IMGD) return;
    int w  = idx % IMGD;  int t = idx / IMGD;
    int h  = t % IMGD;    t /= IMGD;
    int cd = t % IC2;     int b = t / IC2;
    int oc = cd / 28, dz = cd % 28;
    float s = 0.f;
    for (int kd = 0; kd < 3; ++kd)
        for (int kh = 0; kh < 3; ++kh)
            for (int kw = 0; kw < 3; ++kw) {
                int hh = h + kh - 1, ww = w + kw - 1;
                int hc = hh < 0 ? 0 : (hh > 24 ? 24 : hh);
                int wc = ww < 0 ? 0 : (ww > 24 ? 24 : ww);
                float v = img[((size_t)b * 30 + dz + kd) * 625 + hc * 25 + wc];
                bool ok = (hh >= 0) & (hh < 25) & (ww >= 0) & (ww < 25);
                s += (ok ? v : 0.f) * w3[((oc * 3 + kd) * 3 + kh) * 3 + kw];
            }
    s += b3[oc];
    s = (s - mu[oc]) * rsqrtf(var[oc] + EPSf) * g[oc] + bb[oc];
    X1[((size_t)b * IC2 + cd) * XPD2 + (h + 1) * XPD + (w + 1)] = (_Float16)frelu(s);
}

// ---------------------------------------------------------------------------
// conv2d implicit-im2col WMMA GEMM, 16x32 tile per wave.
//   A: direct global b128 fragments from padded weights [32][6048] (reused).
//   B: 32-pixel im2col panel staged in LDS, n-major -> bf = 2x ds_load_b128.
// ---------------------------------------------------------------------------
__global__ __launch_bounds__(256) void k_conv2d_wmma(
    const _Float16* __restrict__ X1, const _Float16* __restrict__ Wh,
    const float* __restrict__ cb, const float* __restrict__ g,
    const float* __restrict__ bb, const float* __restrict__ mu,
    const float* __restrict__ var, float* __restrict__ patches)
{
    __shared__ _Float16 sB[8][1024];           // [wave][32 pixels x 32 k]
    const int lane = threadIdx.x & 31;
    const int wv   = threadIdx.x >> 5;
    int tile = blockIdx.x * 8 + wv;            // B * 40 tiles exactly
    int b   = tile / 40;
    int t2  = tile % 40;
    int mt  = t2 & 1;                          // 2 oc tiles (30 -> 32)
    int nt2 = t2 >> 1;                         // 20 pixel-pair tiles (625 -> 640)
    v8f acc0 = {}, acc1 = {};
    const int mn   = lane & 15;
    const int hsel = lane >> 4;
    const _Float16* aRow = Wh + (size_t)(mt * 16 + mn) * K2 + (hsel << 4);
    for (int it = 0; it < 189; ++it) {         // 6048 / 32 exact
        int k0 = it << 5;
        // A fragment: two contiguous b128 runs, direct from global
        h8 a0 = *(const h8*)(aRow + k0);
        h8 a1 = *(const h8*)(aRow + k0 + 8);
        // B staging: k-row = lane, 32 pixel columns, no masks (halo pad)
        int kk = k0 + lane;
        int ic = kk / 9, rem = kk % 9, kh = rem / 3, kw = rem % 3;
        const _Float16* xBase = X1 + ((size_t)b * IC2 + ic) * XPD2 + kh * XPD + kw;
        #pragma unroll
        for (int j = 0; j < 32; ++j) {
            int p  = nt2 * 32 + j;
            int pc = p < 624 ? p : 624;        // keep reads in-buffer; junk dropped later
            int h  = pc / 25, w = pc % 25;
            sB[wv][j * 32 + lane] = xBase[h * XPD + w];
        }
        __syncthreads();
        h8 b0lo = *(const h8*)&sB[wv][mn * 32 + (hsel << 4)];
        h8 b0hi = *(const h8*)&sB[wv][mn * 32 + (hsel << 4) + 8];
        h8 b1lo = *(const h8*)&sB[wv][(16 + mn) * 32 + (hsel << 4)];
        h8 b1hi = *(const h8*)&sB[wv][(16 + mn) * 32 + (hsel << 4) + 8];
        v16h af  = SHUF16(a0, a1);
        v16h bf0 = SHUF16(b0lo, b0hi);
        v16h bf1 = SHUF16(b1lo, b1hi);
        acc0 = WMMA16(af, bf0, acc0);
        acc1 = WMMA16(af, bf1, acc1);
        __syncthreads();
    }
    // epilogue: bias + BN + ReLU + patch-transpose scatter (both n-subtiles)
    int mOff = hsel << 3;
    #pragma unroll
    for (int rr = 0; rr < 8; ++rr) {
        int oc = mt * 16 + mOff + rr;
        if (oc >= 30) continue;
        float a  = rsqrtf(var[oc] + EPSf) * g[oc];
        float c0 = (cb[oc] - mu[oc]) * a + bb[oc];
        #pragma unroll
        for (int half = 0; half < 2; ++half) {
            int p = nt2 * 32 + half * 16 + mn;
            if (p < 625) {
                float y = frelu((half ? acc1[rr] : acc0[rr]) * a + c0);
                int h = p / 25, w = p % 25;
                int tk = (h / 5) * 5 + (w / 5);
                int f  = ((h % 5) * 5 + (w % 5)) * 30 + oc;
                patches[((size_t)b * Nn + tk) * PATCH + f] = y;
            }
        }
    }
}

// ---------------------------------------------------------------------------
// Generic WMMA GEMM, LDS-free & branch-free, 16x32 tile per wave.
//   A  [M][Kp] fp16 row-major (Kp/32 even); BT [N][Kp] fp16 (B transposed).
//   M % 16 == 0, N % 32 == 0.
// ---------------------------------------------------------------------------
__global__ __launch_bounds__(256) void kg_gemm(
    const _Float16* __restrict__ A, const _Float16* __restrict__ BT,
    const float* __restrict__ bias, const float* __restrict__ residual,
    float* __restrict__ Cf, _Float16* __restrict__ Ch,
    int M, int N, int Kp, int relu)
{
    const int lane = threadIdx.x & 31;
    const int wv   = threadIdx.x >> 5;
    const int mTiles = M >> 4, nTiles2 = N >> 5;
    const int tile = blockIdx.x * 8 + wv;
    if (tile >= mTiles * nTiles2) return;     // no LDS/sync below: safe
    const int mt = tile % mTiles, nt2 = tile / mTiles;
    const int row0 = mt << 4, col0 = nt2 << 5;
    const int mn   = lane & 15;
    const int hsel = lane >> 4;
    const _Float16* aRow  = A  + (size_t)(row0 + mn) * Kp + (hsel << 4);
    const _Float16* bRow0 = BT + (size_t)(col0 + mn) * Kp + (hsel << 4);
    const _Float16* bRow1 = bRow0 + (size_t)16 * Kp;
    v8f acc0 = {}, acc1 = {};
    for (int k0 = 0; k0 < Kp; k0 += 64) {     // 2 K-steps, 4 wmma per iter
        h8 a0  = *(const h8*)(aRow  + k0);
        h8 a1  = *(const h8*)(aRow  + k0 + 8);
        h8 b00 = *(const h8*)(bRow0 + k0);
        h8 b01 = *(const h8*)(bRow0 + k0 + 8);
        h8 b10 = *(const h8*)(bRow1 + k0);
        h8 b11 = *(const h8*)(bRow1 + k0 + 8);
        h8 a2  = *(const h8*)(aRow  + k0 + 32);
        h8 a3  = *(const h8*)(aRow  + k0 + 40);
        h8 b02 = *(const h8*)(bRow0 + k0 + 32);
        h8 b03 = *(const h8*)(bRow0 + k0 + 40);
        h8 b12 = *(const h8*)(bRow1 + k0 + 32);
        h8 b13 = *(const h8*)(bRow1 + k0 + 40);
        if (k0 + 64 < Kp) {                   // speculative L2 prefetch
            __builtin_prefetch(aRow  + k0 + 64, 0, 1);
            __builtin_prefetch(bRow0 + k0 + 64, 0, 1);
            __builtin_prefetch(bRow1 + k0 + 64, 0, 1);
        }
        v16h af0 = SHUF16(a0, a1);
        acc0 = WMMA16(af0, SHUF16(b00, b01), acc0);
        acc1 = WMMA16(af0, SHUF16(b10, b11), acc1);
        v16h af1 = SHUF16(a2, a3);
        acc0 = WMMA16(af1, SHUF16(b02, b03), acc0);
        acc1 = WMMA16(af1, SHUF16(b12, b13), acc1);
    }
    const int mOff = hsel << 3;
    #pragma unroll
    for (int rr = 0; rr < 8; ++rr) {
        int row = row0 + mOff + rr;
        #pragma unroll
        for (int half = 0; half < 2; ++half) {
            int col = col0 + half * 16 + mn;
            float v = half ? acc1[rr] : acc0[rr];
            if (bias)     v += bias[col];
            if (residual) v += residual[(size_t)row * N + col];
            if (relu)     v  = frelu(v);
            if (Cf) Cf[(size_t)row * N + col] = v;
            if (Ch) Ch[(size_t)row * N + col] = (_Float16)v;
        }
    }
}

// ---------------------------------------------------------------------------
// LayerNorm (block per row); writes dpad columns (zero pad) for WMMA K-pad.
// ---------------------------------------------------------------------------
__global__ __launch_bounds__(256) void k_ln(
    const float* __restrict__ in, const float* __restrict__ g,
    const float* __restrict__ bta, const float* __restrict__ pos,
    float* __restrict__ outf, _Float16* __restrict__ outh, int d, int dpad)
{
    __shared__ float red[256];
    int row = blockIdx.x;
    const float* x = in + (size_t)row * d;
    float s = 0.f;
    for (int c = threadIdx.x; c < d; c += 256) s += x[c];
    red[threadIdx.x] = s; __syncthreads();
    for (int o = 128; o > 0; o >>= 1) {
        if ((int)threadIdx.x < o) red[threadIdx.x] += red[threadIdx.x + o];
        __syncthreads();
    }
    float mean = red[0] / d; __syncthreads();
    float s2 = 0.f;
    for (int c = threadIdx.x; c < d; c += 256) { float t = x[c] - mean; s2 += t * t; }
    red[threadIdx.x] = s2; __syncthreads();
    for (int o = 128; o > 0; o >>= 1) {
        if ((int)threadIdx.x < o) red[threadIdx.x] += red[threadIdx.x + o];
        __syncthreads();
    }
    float inv = rsqrtf(red[0] / d + EPSf);
    int tok = row % Nn;
    for (int c = threadIdx.x; c < dpad; c += 256) {
        float v = 0.f;
        if (c < d) {
            v = (x[c] - mean) * inv * g[c] + bta[c];
            if (pos) v += pos[tok * d + c];
        }
        if (outf) outf[(size_t)row * dpad + c] = v;
        if (outh) outh[(size_t)row * dpad + c] = (_Float16)v;
    }
}

// ---------------------------------------------------------------------------
// Region-masked attention (4 quadrant masks), N=25, DH=64.  One block / (b,h).
// q/k/v tiles loaded via async global->LDS (ASYNCcnt).
// ---------------------------------------------------------------------------
static __device__ __forceinline__ bool region_mask(int rg, int ii, int ij, int j) {
    int ji = j / 5, jj = j % 5;
    switch (rg) {
        case 0:  return (ji <= ii) && (jj <= ij);
        case 1:  return (ji <= ii) && (jj >= ij);
        case 2:  return (ji >= ii) && (jj <= ij);
        default: return (ji >= ii) && (jj >= ij);
    }
}

__global__ __launch_bounds__(256) void k_attn1(
    const float* __restrict__ QKV, _Float16* __restrict__ AO)
{
    __shared__ float q[Nn * DHh], k[Nn * DHh], v[Nn * DHh];
    __shared__ float dots[Nn * Nn];
    __shared__ float attn[4 * Nn * Nn];
    int bh = blockIdx.x, b = bh >> 3, h = bh & 7;
    const float* base = QKV + (size_t)(b * Nn) * (3 * INNER);
    for (int t = threadIdx.x; t < Nn * DHh; t += 256) {
        int i = t >> 6, d = t & 63;
        async_ld32(&q[t], base + i * 1536 + h * 64 + d);
        async_ld32(&k[t], base + i * 1536 + 512 + h * 64 + d);
        async_ld32(&v[t], base + i * 1536 + 1024 + h * 64 + d);
    }
    async_wait0();
    __syncthreads();
    for (int t = threadIdx.x; t < 625; t += 256) {
        int i = t / 25, j = t % 25;
        float s = 0.f;
        for (int d = 0; d < 64; ++d) s += q[i * 64 + d] * k[j * 64 + d];
        dots[t] = s * 0.125f;                  // DH^-0.5
    }
    __syncthreads();
    for (int t = threadIdx.x; t < 100; t += 256) {
        int rg = t / 25, i = t % 25;
        int ii = i / 5, ij = i % 5;
        float mx = -1e30f;
        for (int j = 0; j < 25; ++j)
            if (region_mask(rg, ii, ij, j)) mx = fmaxf(mx, dots[i * 25 + j]);
        float sum = 0.f;
        for (int j = 0; j < 25; ++j)
            if (region_mask(rg, ii, ij, j)) sum += __expf(dots[i * 25 + j] - mx);
        float invs = 1.f / sum;
        for (int j = 0; j < 25; ++j)
            attn[(rg * 25 + i) * 25 + j] =
                region_mask(rg, ii, ij, j) ? frelu(__expf(dots[i * 25 + j] - mx) * invs) : 0.f;
    }
    __syncthreads();
    for (int t = threadIdx.x; t < Nn * DHh; t += 256) {
        int i = t >> 6, d = t & 63;
        float s = 0.f;
        for (int rg = 0; rg < 4; ++rg)
            for (int j = 0; j < 25; ++j)
                s += attn[(rg * 25 + i) * 25 + j] * v[j * 64 + d];
        AO[(size_t)(b * Nn + i) * INNER + h * 64 + d] = (_Float16)(s * 0.25f);
    }
}

// ---------------------------------------------------------------------------
// Plain MHA, N=25, dh=16.  One block / (b,h).  Async q/k/v loads.
// ---------------------------------------------------------------------------
__global__ __launch_bounds__(256) void k_attn2(
    const float* __restrict__ QKV2, _Float16* __restrict__ O2)
{
    __shared__ float q[Nn * 16], k[Nn * 16], v[Nn * 16], att[Nn * Nn];
    int bh = blockIdx.x, b = bh >> 3, h = bh & 7;
    const float* base = QKV2 + (size_t)(b * Nn) * 384;
    for (int t = threadIdx.x; t < 400; t += 256) {
        int i = t >> 4, d = t & 15;
        async_ld32(&q[t], base + i * 384 + h * 16 + d);
        async_ld32(&k[t], base + i * 384 + 128 + h * 16 + d);
        async_ld32(&v[t], base + i * 384 + 256 + h * 16 + d);
    }
    async_wait0();
    __syncthreads();
    for (int t = threadIdx.x; t < 625; t += 256) {
        int i = t / 25, j = t % 25;
        float s = 0.f;
        for (int d = 0; d < 16; ++d) s += q[i * 16 + d] * k[j * 16 + d];
        att[t] = s * 0.25f;                    // 16^-0.5
    }
    __syncthreads();
    for (int i = threadIdx.x; i < 25; i += 256) {
        float mx = -1e30f;
        for (int j = 0; j < 25; ++j) mx = fmaxf(mx, att[i * 25 + j]);
        float sum = 0.f;
        for (int j = 0; j < 25; ++j) sum += __expf(att[i * 25 + j] - mx);
        float invs = 1.f / sum;
        for (int j = 0; j < 25; ++j) att[i * 25 + j] = __expf(att[i * 25 + j] - mx) * invs;
    }
    __syncthreads();
    for (int t = threadIdx.x; t < 400; t += 256) {
        int i = t >> 4, d = t & 15;
        float s = 0.f;
        for (int j = 0; j < 25; ++j) s += att[i * 25 + j] * v[j * 16 + d];
        O2[(size_t)(b * Nn + i) * DIMd + h * 16 + d] = (_Float16)s;
    }
}

// ---------------------------------------------------------------------------
// bilinear 5->25 resize (half-pixel) + ReLU + seg head (128->16)
// ---------------------------------------------------------------------------
__global__ __launch_bounds__(128) void k_seg(
    const float* __restrict__ x4, const float* __restrict__ segw,
    const float* __restrict__ segb, float* __restrict__ out)
{
    __shared__ float sh[128];
    int id = blockIdx.x;
    int b = id / 625, p = id % 625;
    int oh = p / 25, ow = p % 25;
    int d = threadIdx.x;
    float fy = (oh + 0.5f) * 0.2f - 0.5f;
    float fx = (ow + 0.5f) * 0.2f - 0.5f;
    int y0 = (int)floorf(fy); float wy = fy - (float)y0;
    int x0 = (int)floorf(fx); float wx = fx - (float)x0;
    int y0c = y0 < 0 ? 0 : (y0 > 4 ? 4 : y0);
    int y1c = (y0 + 1) < 0 ? 0 : ((y0 + 1) > 4 ? 4 : y0 + 1);
    int x0c = x0 < 0 ? 0 : (x0 > 4 ? 4 : x0);
    int x1c = (x0 + 1) < 0 ? 0 : ((x0 + 1) > 4 ? 4 : x0 + 1);
    auto pix = [&](int yy, int xx) -> float {
        int fi = d * 25 + yy * 5 + xx;           // reshape(b, DIM, 5, 5) reinterpret
        return x4[(size_t)(b * Nn + (fi >> 7)) * DIMd + (fi & 127)];
    };
    float v = (1.f - wy) * ((1.f - wx) * pix(y0c, x0c) + wx * pix(y0c, x1c))
            +        wy  * ((1.f - wx) * pix(y1c, x0c) + wx * pix(y1c, x1c));
    sh[d] = frelu(v);
    __syncthreads();
    if (d < NCc) {
        float s = segb[d];
        for (int c = 0; c < 128; ++c) s += sh[c] * segw[c * NCc + d];
        out[((size_t)b * NCc + d) * 625 + p] = s;
    }
}

// ---------------------------------------------------------------------------
// cls head: center pixel -> relu(x@w1+b1)@w2+b2
// ---------------------------------------------------------------------------
__global__ __launch_bounds__(32) void k_cls(
    const float* __restrict__ seg, const float* __restrict__ w1,
    const float* __restrict__ b1, const float* __restrict__ w2,
    const float* __restrict__ b2, float* __restrict__ out)
{
    __shared__ float v[NCc], h1[NCc];
    int b = blockIdx.x, t = threadIdx.x;
    if (t < NCc) v[t] = seg[((size_t)b * NCc + t) * 625 + 12 * 25 + 12];
    __syncthreads();
    if (t < NCc) {
        float s = b1[t];
        for (int k = 0; k < NCc; ++k) s += v[k] * w1[k * NCc + t];
        h1[t] = frelu(s);
    }
    __syncthreads();
    if (t < NCc) {
        float s = b2[t];
        for (int k = 0; k < NCc; ++k) s += h1[k] * w2[k * NCc + t];
        out[(size_t)b * NCc + t] = s;
    }
}

// ---------------------------------------------------------------------------
// host orchestration
// ---------------------------------------------------------------------------
extern "C" void kernel_launch(void* const* d_in, const int* in_sizes, int n_in,
                              void* d_out, int out_size, void* d_ws, size_t ws_size,
                              hipStream_t stream)
{
    (void)in_sizes; (void)n_in; (void)out_size; (void)ws_size;
    const float* img      = (const float*)d_in[0];
    const float* conv3d_w = (const float*)d_in[2];
    const float* conv3d_b = (const float*)d_in[3];
    const float* bn3d_g   = (const float*)d_in[4];
    const float* bn3d_b   = (const float*)d_in[5];
    const float* bn3d_m   = (const float*)d_in[6];
    const float* bn3d_v   = (const float*)d_in[7];
    const float* conv2d_w = (const float*)d_in[8];
    const float* conv2d_b = (const float*)d_in[9];
    const float* bn2d_g   = (const float*)d_in[10];
    const float* bn2d_b   = (const float*)d_in[11];
    const float* bn2d_m   = (const float*)d_in[12];
    const float* bn2d_v   = (const float*)d_in[13];
    const float* lnp_g    = (const float*)d_in[14];
    const float* lnp_b    = (const float*)d_in[15];
    const float* emb_w    = (const float*)d_in[16];
    const float* emb_b    = (const float*)d_in[17];
    const float* lne_g    = (const float*)d_in[18];
    const float* lne_b    = (const float*)d_in[19];
    const float* pos_emb  = (const float*)d_in[20];
    const float* aln_g    = (const float*)d_in[21];
    const float* aln_b    = (const float*)d_in[22];
    const float* qkv_w    = (const float*)d_in[23];
    const float* qkv_b    = (const float*)d_in[24];
    const float* aout_w   = (const float*)d_in[25];
    const float* aout_b   = (const float*)d_in[26];
    const float* fln_g    = (const float*)d_in[27];
    const float* fln_b    = (const float*)d_in[28];
    const float* ff_w1    = (const float*)d_in[29];
    const float* ff_b1    = (const float*)d_in[30];
    const float* ff_w2    = (const float*)d_in[31];
    const float* ff_b2    = (const float*)d_in[32];
    const float* mha_in_w = (const float*)d_in[33];
    const float* mha_in_b = (const float*)d_in[34];
    const float* mha_out_w= (const float*)d_in[35];
    const float* mha_out_b= (const float*)d_in[36];
    const float* nln_g    = (const float*)d_in[37];
    const float* nln_b    = (const float*)d_in[38];
    const float* seg_w    = (const float*)d_in[39];
    const float* seg_b    = (const float*)d_in[40];
    const float* hd_w1    = (const float*)d_in[41];
    const float* hd_b1    = (const float*)d_in[42];
    const float* hd_w2    = (const float*)d_in[43];
    const float* hd_b2    = (const float*)d_in[44];

    float* out_seg = (float*)d_out;                    // [128][16][25][25]
    float* out_cls = (float*)d_out + 1280000;          // [128][16]

    // bump allocator over workspace
    char* wsb = (char*)d_ws;
    size_t off = 0;
    auto alloc = [&](size_t bytes) -> void* {
        off = (off + 255) & ~(size_t)255;
        void* p = wsb + off;
        off += bytes;
        return p;
    };
    _Float16* X1h     = (_Float16*)alloc((size_t)Bsz * IC2 * XPD2 * 2);
    _Float16* Wc2h    = (_Float16*)alloc((size_t)32 * K2 * 2);
    float*    patches = (float*)   alloc((size_t)TOK * PATCH * 4);
    _Float16* A0h     = (_Float16*)alloc((size_t)TOK * PATCHP * 2);
    _Float16* emb_wt  = (_Float16*)alloc((size_t)DIMd * PATCHP * 2);
    float*    E       = (float*)   alloc((size_t)TOK * DIMd * 4);
    float*    x       = (float*)   alloc((size_t)TOK * DIMd * 4);
    _Float16* xnh     = (_Float16*)alloc((size_t)TOK * DIMd * 2);
    _Float16* qkv_wt  = (_Float16*)alloc((size_t)(3 * INNER) * DIMd * 2);
    float*    QKV     = (float*)   alloc((size_t)TOK * 3 * INNER * 4);
    _Float16* AOh     = (_Float16*)alloc((size_t)TOK * INNER * 2);
    _Float16* aout_wt = (_Float16*)alloc((size_t)DIMd * INNER * 2);
    float*    x2      = (float*)   alloc((size_t)TOK * DIMd * 4);
    _Float16* fh      = (_Float16*)alloc((size_t)TOK * DIMd * 2);
    _Float16* ff1_wt  = (_Float16*)alloc((size_t)MLPd * DIMd * 2);
    _Float16* h1h     = (_Float16*)alloc((size_t)TOK * MLPd * 2);
    _Float16* ff2_wt  = (_Float16*)alloc((size_t)DIMd * MLPd * 2);
    float*    x3      = (float*)   alloc((size_t)TOK * DIMd * 4);
    _Float16* x3h     = (_Float16*)alloc((size_t)TOK * DIMd * 2);
    _Float16* mi_wt   = (_Float16*)alloc((size_t)(3 * DIMd) * DIMd * 2);
    float*    QKV2    = (float*)   alloc((size_t)TOK * 3 * DIMd * 4);
    _Float16* O2h     = (_Float16*)alloc((size_t)TOK * DIMd * 2);
    _Float16* mo_wt   = (_Float16*)alloc((size_t)DIMd * DIMd * 2);
    float*    M2      = (float*)   alloc((size_t)TOK * DIMd * 4);
    float*    x4      = (float*)   alloc((size_t)TOK * DIMd * 4);

    auto f2ht = [&](const float* W, _Float16* BT, int K, int N, int Kp) {
        int n = N * Kp;
        k_f2h_t<<<(n + 255) / 256, 256, 0, stream>>>(W, BT, K, N, Kp);
    };
    auto gemm = [&](const _Float16* A, const _Float16* BT, const float* bias,
                    const float* res, float* Cf, _Float16* Ch,
                    int M, int N, int Kp, int relu) {
        int tiles = (M / 16) * (N / 32);
        kg_gemm<<<(tiles + 7) / 8, 256, 0, stream>>>(A, BT, bias, res, Cf, Ch, M, N, Kp, relu);
    };

    // weight conversions
    {
        int n = 32 * K2;
        k_prep_w2<<<(n + 255) / 256, 256, 0, stream>>>(conv2d_w, Wc2h);
    }
    f2ht(emb_w,     emb_wt, PATCH, DIMd, PATCHP);
    f2ht(qkv_w,     qkv_wt, DIMd, 3 * INNER, DIMd);
    f2ht(aout_w,    aout_wt, INNER, DIMd, INNER);
    f2ht(ff_w1,     ff1_wt, DIMd, MLPd, DIMd);
    f2ht(ff_w2,     ff2_wt, MLPd, DIMd, MLPd);
    f2ht(mha_in_w,  mi_wt, DIMd, 3 * DIMd, DIMd);
    f2ht(mha_out_w, mo_wt, DIMd, DIMd, DIMd);

    // zero halo-padded conv3d output, then fill interior
    {
        int n8 = (Bsz * IC2 * XPD2) / 8;      // divisible by 8
        k_zero_h<<<(n8 + 255) / 256, 256, 0, stream>>>(X1h, n8);
    }
    k_conv3d<<<(Bsz * IC2 * 625 + 255) / 256, 256, 0, stream>>>(
        img, conv3d_w, conv3d_b, bn3d_g, bn3d_b, bn3d_m, bn3d_v, X1h);

    // conv2d implicit-im2col WMMA + BN + ReLU + patchify (16x32 tiles)
    k_conv2d_wmma<<<(Bsz * 40) / 8, 256, 0, stream>>>(
        X1h, Wc2h, conv2d_b, bn2d_g, bn2d_b, bn2d_m, bn2d_v, patches);

    // LN(lnp) -> fp16 tokens (K-padded to 768)
    k_ln<<<TOK, 256, 0, stream>>>(patches, lnp_g, lnp_b, nullptr, nullptr, A0h, PATCH, PATCHP);
    // embed
    gemm(A0h, emb_wt, emb_b, nullptr, E, nullptr, TOK, DIMd, PATCHP, 0);
    // LN(lne) + pos
    k_ln<<<TOK, 256, 0, stream>>>(E, lne_g, lne_b, pos_emb, x, nullptr, DIMd, DIMd);
    // LN(aln) -> fp16
    k_ln<<<TOK, 256, 0, stream>>>(x, aln_g, aln_b, nullptr, nullptr, xnh, DIMd, DIMd);
    // qkv
    gemm(xnh, qkv_wt, qkv_b, nullptr, QKV, nullptr, TOK, 3 * INNER, DIMd, 0);
    // region attention
    k_attn1<<<Bsz * HEADS, 256, 0, stream>>>(QKV, AOh);
    // attn out + residual
    gemm(AOh, aout_wt, aout_b, x, x2, nullptr, TOK, DIMd, INNER, 0);
    // FFN
    k_ln<<<TOK, 256, 0, stream>>>(x2, fln_g, fln_b, nullptr, nullptr, fh, DIMd, DIMd);
    gemm(fh, ff1_wt, ff_b1, nullptr, nullptr, h1h, TOK, MLPd, DIMd, 1);
    gemm(h1h, ff2_wt, ff_b2, x2, x3, x3h, TOK, DIMd, MLPd, 0);
    // MHA2
    gemm(x3h, mi_wt, mha_in_b, nullptr, QKV2, nullptr, TOK, 3 * DIMd, DIMd, 0);
    k_attn2<<<Bsz * HEADS, 256, 0, stream>>>(QKV2, O2h);
    gemm(O2h, mo_wt, mha_out_b, nullptr, M2, nullptr, TOK, DIMd, DIMd, 0);
    // LN(nln)
    k_ln<<<TOK, 256, 0, stream>>>(M2, nln_g, nln_b, nullptr, x4, nullptr, DIMd, DIMd);
    // resize + relu + seg head
    k_seg<<<Bsz * 625, 128, 0, stream>>>(x4, seg_w, seg_b, out_seg);
    // cls head
    k_cls<<<Bsz, 32, 0, stream>>>(out_seg, hd_w1, hd_b1, hd_w2, hd_b2, out_cls);
}
